// MultiHeadedAttention_19146964205644
// MI455X (gfx1250) — matrix operand order repaired and checked by reading, not compiled, not run on hardware
//
#include <hip/hip_runtime.h>
#include <hip/hip_bf16.h>

// Problem constants (from reference): B=4, T=2048, D=384, H=4, DK=96
#define B_  4
#define T_  2048
#define D_  384
#define H_  4
#define DK_ 96
#define BT_ (B_ * T_)   // 8192 rows
#define BH_ (B_ * H_)   // 16 (batch*heads)

// ---- CDNA5 async Global->LDS path (ASYNCcnt), probe-guarded -----------------
#if defined(__has_builtin)
#  if __has_builtin(__builtin_amdgcn_global_load_async_to_lds_b128)
#    define HAVE_ASYNC_COPY 1
#  endif
#endif
#ifndef HAVE_ASYNC_COPY
#  define HAVE_ASYNC_COPY 0
#endif

#if HAVE_ASYNC_COPY
#  if __has_builtin(__builtin_amdgcn_s_wait_asynccnt)
#    define WAIT_ASYNC(n) __builtin_amdgcn_s_wait_asynccnt(n)
#  else
#    define WAIT_ASYNC(n) asm volatile("s_wait_asynccnt " #n ::: "memory")
#  endif
#else
#  define WAIT_ASYNC(n) do {} while (0)
#endif

typedef __attribute__((ext_vector_type(16))) __bf16 v16bf;
typedef __attribute__((ext_vector_type(8)))  float  v8f;

// GNU-vector int4: exact pointee type expected by the async-load builtin
// (diagnostic showed 'int __attribute__((vector_size(4*sizeof(int)))) __device__ *').
typedef int vsi4 __attribute__((vector_size(16)));
typedef __attribute__((address_space(1))) vsi4 as1_vsi4;   // global
typedef __attribute__((address_space(3))) vsi4 as3_vsi4;   // LDS

union ABOp { unsigned int u[8]; v16bf v; };

__device__ __forceinline__ unsigned short f2bf(float f) {
  unsigned int u = __float_as_uint(f);
  u += 0x7FFFu + ((u >> 16) & 1u);       // round-to-nearest-even
  return (unsigned short)(u >> 16);
}

// 16B Global->LDS copy. Async variant uses GLOBAL_LOAD_ASYNC_TO_LDS_B128:
// data lands in LDS without touching VGPRs, tracked by ASYNCcnt.
// Generic->AS1 is numerically identity on amdgcn; generic LDS pointers carry
// the LDS byte offset in the low 32 bits (aperture rule), so the integer casts
// below are exact addrspacecasts (and also drop const/retype the pointee).
__device__ __forceinline__ void copy16(const unsigned short* g, unsigned short* l) {
#if HAVE_ASYNC_COPY
  __builtin_amdgcn_global_load_async_to_lds_b128(
      (as1_vsi4*)(unsigned long long)g,
      (as3_vsi4*)(unsigned int)(unsigned long long)l,
      0, 0);
#else
  *(uint4*)l = *(const uint4*)g;
#endif
}

// A operand (16x32 bf16, MxK): lane m=lane&15, half=lane>>4.
// => two contiguous 16B chunks at dword offsets {4*half} and {8 + 4*half} of row m.
__device__ __forceinline__ void load_a(const unsigned int* row, int half, ABOp& op) {
  uint4 lo = *(const uint4*)(row + 4 * half);
  uint4 hi = *(const uint4*)(row + 8 + 4 * half);
  op.u[0] = lo.x; op.u[1] = lo.y; op.u[2] = lo.z; op.u[3] = lo.w;
  op.u[4] = hi.x; op.u[5] = hi.y; op.u[6] = hi.z; op.u[7] = hi.w;
}

// B operand (32x16 bf16, KxN): lane n=lane&15 is the output column; same K order
// as A => two 16B chunks at dword offsets {4*half} and {8 + 4*half} of the
// column's K-vector (row of the transposed-B source).
__device__ __forceinline__ void load_b(const unsigned int* row, int half, ABOp& op) {
  uint4 lo = *(const uint4*)(row + 4 * half);
  uint4 hi = *(const uint4*)(row + 8 + 4 * half);
  op.u[0] = lo.x; op.u[1] = lo.y; op.u[2] = lo.z; op.u[3] = lo.w;
  op.u[4] = hi.x; op.u[5] = hi.y; op.u[6] = hi.z; op.u[7] = hi.w;
}

__device__ __forceinline__ v8f wmma_bf16(const ABOp& a, const ABOp& b, v8f c) {
  return __builtin_amdgcn_wmma_f32_16x16x32_bf16(
      /*neg_a=*/false, a.v, /*neg_b=*/false, b.v,
      /*c_mod=*/(short)0, c, /*reuse_a=*/false, /*reuse_b=*/false);
}

// ---------------------------------------------------------------- f32 -> bf16
__global__ void __launch_bounds__(256)
cvt_bf16(const float* __restrict__ x, unsigned short* __restrict__ y, int n) {
  int i = blockIdx.x * 256 + threadIdx.x;
  if (i < n) y[i] = f2bf(x[i]);
}

// ------------------------------------------------- GEMM: Y[M,384] = X * W^T + b
// One wave computes a 16(M) x 64(N) strip; A operand shared by 4 WMMAs per K-step.
__global__ void __launch_bounds__(256)
gemm_bf16nt(const unsigned short* __restrict__ X,
            const unsigned short* __restrict__ W,
            const float* __restrict__ bias,
            float* __restrict__ Y, int M) {
  const int lane = threadIdx.x & 31;
  const int wave = threadIdx.x >> 5;
  const int half = lane >> 4;
  const int lm   = lane & 15;
  const int wid  = blockIdx.x * 8 + wave;
  const int nStrips = D_ / 64;                // 6
  const int mTile  = wid / nStrips;
  const int nStrip = wid % nStrips;
  if (mTile * 16 >= M) return;

  const unsigned int* Xu = (const unsigned int*)X;
  const unsigned int* Wu = (const unsigned int*)W;
  const unsigned int* xrow = Xu + (size_t)(mTile * 16 + lm) * (D_ / 2);
  const int nb = nStrip * 64;
  const unsigned int* w0 = Wu + (size_t)(nb +  0 + lm) * (D_ / 2);
  const unsigned int* w1 = Wu + (size_t)(nb + 16 + lm) * (D_ / 2);
  const unsigned int* w2 = Wu + (size_t)(nb + 32 + lm) * (D_ / 2);
  const unsigned int* w3 = Wu + (size_t)(nb + 48 + lm) * (D_ / 2);

  v8f a0 = {}, a1 = {}, a2 = {}, a3 = {};
#pragma unroll
  for (int ks = 0; ks < D_ / 32; ++ks) {      // 12 K-steps of 32
    ABOp av, b0, b1, b2, b3;
    load_a(xrow + ks * 16, half, av);
    load_b(w0 + ks * 16, half, b0);
    load_b(w1 + ks * 16, half, b1);
    load_b(w2 + ks * 16, half, b2);
    load_b(w3 + ks * 16, half, b3);
    a0 = wmma_bf16(av, b0, a0);
    a1 = wmma_bf16(av, b1, a1);
    a2 = wmma_bf16(av, b2, a2);
    a3 = wmma_bf16(av, b3, a3);
  }
  const float bb0 = bias[nb + lm];
  const float bb1 = bias[nb + 16 + lm];
  const float bb2 = bias[nb + 32 + lm];
  const float bb3 = bias[nb + 48 + lm];
#pragma unroll
  for (int v = 0; v < 8; ++v) {
    int r = mTile * 16 + 8 * half + v;        // C layout: row v + 8*(lane>>4)
    float* yr = Y + (size_t)r * D_ + nb;
    yr[lm]      = a0[v] + bb0;
    yr[lm + 16] = a1[v] + bb1;
    yr[lm + 32] = a2[v] + bb2;
    yr[lm + 48] = a3[v] + bb3;
  }
}

// ---------------------------------- per-head LayerNorm (dk=96, eps=1e-12) -> bf16
__global__ void __launch_bounds__(256)
ln_head(const float* __restrict__ Xf, const float* __restrict__ w,
        const float* __restrict__ bvec, unsigned short* __restrict__ O) {
  const int lane = threadIdx.x & 31;
  const int row  = blockIdx.x * 8 + (threadIdx.x >> 5);  // (b*H+h)*T + t
  const int t   = row % T_;
  const int bhh = row / T_;
  const int bq  = bhh / H_;
  const int hq  = bhh % H_;
  const float* xr = Xf + ((size_t)(bq * T_ + t)) * D_ + hq * DK_;
  float x0 = xr[lane], x1 = xr[lane + 32], x2 = xr[lane + 64];
  float s = x0 + x1 + x2;
  float q = x0 * x0 + x1 * x1 + x2 * x2;
#pragma unroll
  for (int m = 1; m < 32; m <<= 1) {
    s += __shfl_xor(s, m, 32);
    q += __shfl_xor(q, m, 32);
  }
  const float mu  = s * (1.0f / 96.0f);
  const float var = q * (1.0f / 96.0f) - mu * mu;
  const float rs  = rsqrtf(var + 1e-12f);
  unsigned short* orow = O + (size_t)row * DK_;
  orow[lane]      = f2bf((x0 - mu) * rs * w[lane]      + bvec[lane]);
  orow[lane + 32] = f2bf((x1 - mu) * rs * w[lane + 32] + bvec[lane + 32]);
  orow[lane + 64] = f2bf((x2 - mu) * rs * w[lane + 64] + bvec[lane + 64]);
}

// -------- V: (B,T,D) f32 -> transposed bf16 (B,H,DK,T) so PV B-operands are contiguous
__global__ void __launch_bounds__(256)
v_split(const float* __restrict__ Vf, unsigned short* __restrict__ Vt) {
  int i = blockIdx.x * 256 + threadIdx.x;
  if (i >= BT_ * D_) return;
  int d  = i % D_;
  int bt = i / D_;
  int t  = bt % T_;
  int b  = bt / T_;
  int h = d / DK_, dd = d % DK_;
  Vt[((size_t)(b * H_ + h) * DK_ + dd) * T_ + t] = f2bf(Vf[i]);
}

// ------------------------------------------------------------- flash attention
// Grid: (T/64, B*H). 128 threads = 4 waves; each wave owns a 16-row Q tile.
// K/V^T tiles are double-buffered in LDS and prefetched with async
// Global->LDS loads (ASYNCcnt) while the current tile's 12 WMMAs + softmax run.
__global__ void __launch_bounds__(128)
attn_kernel(const unsigned short* __restrict__ Qn,   // (B*H, T, 96) bf16
            const unsigned short* __restrict__ Kn,   // (B*H, T, 96) bf16
            const unsigned short* __restrict__ Vt,   // (B*H, 96, T) bf16
            const int* __restrict__ mask,            // (B, T)
            unsigned short* __restrict__ ctx) {      // (B, T, D) bf16, heads merged
  __shared__ __align__(16) unsigned short kbuf[2][32 * DK_];   // 32 keys x 96
  __shared__ __align__(16) unsigned short vbuf[2][DK_ * 32];   // 96 x 32 keys
  __shared__ __align__(16) unsigned short pbuf[4][16 * 32];    // per-wave P scratch

  const int tid  = threadIdx.x;
  const int lane = tid & 31;
  const int wave = tid >> 5;
  const int half = lane >> 4;
  const int lm   = lane & 15;
  const int bh   = blockIdx.y;
  const int b    = bh / H_;
  const int h    = bh % H_;
  const int qBase = blockIdx.x * 64 + wave * 16;

  // Q tile resident in registers: 3 A-operands (dk chunks of 32)
  const unsigned int* Qu = (const unsigned int*)(Qn + (size_t)bh * T_ * DK_);
  ABOp qa0, qa1, qa2;
  {
    const unsigned int* qrow = Qu + (size_t)(qBase + lm) * (DK_ / 2);
    load_a(qrow +  0, half, qa0);
    load_a(qrow + 16, half, qa1);
    load_a(qrow + 32, half, qa2);
  }

  v8f o0 = {}, o1 = {}, o2 = {}, o3 = {}, o4 = {}, o5 = {};
  float mrow[8], lrow[8];
#pragma unroll
  for (int v = 0; v < 8; ++v) { mrow[v] = -3.0e38f; lrow[v] = 0.0f; }

  const unsigned short* Kg = Kn + (size_t)bh * T_ * DK_;
  const unsigned short* Vg = Vt + (size_t)bh * DK_ * T_;
  const int   nIter = T_ / 32;                       // 64
  const float scale = 0.1020620726159657f;           // 1/sqrt(96)
  const float LOG2E = 1.4426950408889634f;

  // Stage one 32-key tile into buffer sb: 3 + 3 b128 chunks per thread
  // (6 async-load instructions per wave => ASYNCcnt quantum of 6).
  auto stage = [&](int sb, int it2) {
    const unsigned short* kg = Kg + (size_t)it2 * 32 * DK_;   // contiguous 6 KB
#pragma unroll
    for (int c = 0; c < 3; ++c) {
      int ch = tid + c * 128;                                 // 384 chunks
      copy16(kg + ch * 8, &kbuf[sb][ch * 8]);
    }
#pragma unroll
    for (int c = 0; c < 3; ++c) {
      int ch = tid + c * 128;                                 // 4 chunks / V^T row
      int d = ch >> 2, part = ch & 3;
      copy16(Vg + (size_t)d * T_ + it2 * 32 + part * 8,
             &vbuf[sb][d * 32 + part * 8]);
    }
  };

  stage(0, 0);
  for (int it = 0; it < nIter; ++it) {
    const int cb = it & 1;
    if (it + 1 < nIter) {
      stage(cb ^ 1, it + 1);   // prefetch next tile into the other buffer
      WAIT_ASYNC(6);           // current tile's 6 loads retired (in-order)
    } else {
      WAIT_ASYNC(0);
    }
    __syncthreads();           // publish all waves' staging

    // ---- scores: c0 = keys 0-15, c1 = keys 16-31 (16x16 f32 tiles)
    const unsigned int* ku = (const unsigned int*)kbuf[cb];
    const unsigned int* vu = (const unsigned int*)vbuf[cb];
    v8f c0 = {}, c1 = {};
#pragma unroll
    for (int kc = 0; kc < 3; ++kc) {
      ABOp b0, b1;
      load_b(ku + (size_t)lm * (DK_ / 2) + kc * 16, half, b0);
      load_b(ku + (size_t)(lm + 16) * (DK_ / 2) + kc * 16, half, b1);
      const ABOp& aop = (kc == 0) ? qa0 : ((kc == 1) ? qa1 : qa2);
      c0 = wmma_bf16(aop, b0, c0);
      c1 = wmma_bf16(aop, b1, c1);
    }

    const int  key0 = it * 32 + lm;
    const bool msk0 = (mask[b * T_ + key0] == 0);
    const bool msk1 = (mask[b * T_ + key0 + 16] == 0);

    // ---- online softmax. C layout: element v is (row v+8*half, col lm);
    //      row stats reduce over 16 lanes per half (xor masks 1,2,4,8).
    unsigned short* pb = pbuf[wave];
    float alpha[8];
#pragma unroll
    for (int v = 0; v < 8; ++v) {
      float sa = msk0 ? -3.0e38f : c0[v] * scale;
      float sb = msk1 ? -3.0e38f : c1[v] * scale;
      float mx = fmaxf(sa, sb);
      mx = fmaxf(mx, __shfl_xor(mx, 1, 32));
      mx = fmaxf(mx, __shfl_xor(mx, 2, 32));
      mx = fmaxf(mx, __shfl_xor(mx, 4, 32));
      mx = fmaxf(mx, __shfl_xor(mx, 8, 32));
      float mnew = fmaxf(mrow[v], mx);
      float a = exp2f((mrow[v] - mnew) * LOG2E);
      mrow[v] = mnew;
      float p0 = msk0 ? 0.0f : exp2f((sa - mnew) * LOG2E);
      float p1 = msk1 ? 0.0f : exp2f((sb - mnew) * LOG2E);
      float rsum = p0 + p1;
      rsum += __shfl_xor(rsum, 1, 32);
      rsum += __shfl_xor(rsum, 2, 32);
      rsum += __shfl_xor(rsum, 4, 32);
      rsum += __shfl_xor(rsum, 8, 32);
      lrow[v] = lrow[v] * a + rsum;
      alpha[v] = a;
      // C->A transpose of P through per-wave LDS scratch (same-wave DS in-order)
      pb[(v + 8 * half) * 32 + lm]      = f2bf(p0);
      pb[(v + 8 * half) * 32 + lm + 16] = f2bf(p1);
    }
#pragma unroll
    for (int v = 0; v < 8; ++v) {
      o0[v] *= alpha[v]; o1[v] *= alpha[v]; o2[v] *= alpha[v];
      o3[v] *= alpha[v]; o4[v] *= alpha[v]; o5[v] *= alpha[v];
    }

    // ---- O += P (16x32) x V (32x16 per dk block)
    ABOp pa;
    load_a((const unsigned int*)pb + lm * 16, half, pa);
#pragma unroll
    for (int j = 0; j < 6; ++j) {
      ABOp vb;
      load_b(vu + (size_t)(j * 16 + lm) * 16, half, vb);
      v8f& oj = (j == 0) ? o0 : (j == 1) ? o1 : (j == 2) ? o2
              : (j == 3) ? o3 : (j == 4) ? o4 : o5;
      oj = wmma_bf16(pa, vb, oj);
    }
    __syncthreads();   // all waves done reading tile cb before it is re-staged
  }

  // ---- normalize + store merged-head context (guard fully-masked rows -> 0)
#pragma unroll
  for (int v = 0; v < 8; ++v) {
    float inv = (lrow[v] > 0.0f) ? (1.0f / lrow[v]) : 0.0f;
    int t = qBase + v + 8 * half;
    unsigned short* cr = ctx + ((size_t)(b * T_ + t)) * D_ + h * DK_;
    cr[lm]      = f2bf(o0[v] * inv);
    cr[lm + 16] = f2bf(o1[v] * inv);
    cr[lm + 32] = f2bf(o2[v] * inv);
    cr[lm + 48] = f2bf(o3[v] * inv);
    cr[lm + 64] = f2bf(o4[v] * inv);
    cr[lm + 80] = f2bf(o5[v] * inv);
  }
}

// ---------------------------------------------------------------------- launch
extern "C" void kernel_launch(void* const* d_in, const int* in_sizes, int n_in,
                              void* d_out, int out_size, void* d_ws, size_t ws_size,
                              hipStream_t stream) {
  const float* query = (const float*)d_in[0];
  const float* key   = (const float*)d_in[1];
  const float* value = (const float*)d_in[2];
  const int*   mask  = (const int*)d_in[3];
  const float* Wq = (const float*)d_in[4];   const float* bq = (const float*)d_in[5];
  const float* Wk = (const float*)d_in[6];   const float* bk = (const float*)d_in[7];
  const float* Wv = (const float*)d_in[8];   const float* bv = (const float*)d_in[9];
  const float* Wo = (const float*)d_in[10];  const float* bo = (const float*)d_in[11];
  const float* qn_w = (const float*)d_in[12]; const float* qn_b = (const float*)d_in[13];
  const float* kn_w = (const float*)d_in[14]; const float* kn_b = (const float*)d_in[15];
  float* out = (float*)d_out;

  char* ws = (char*)d_ws;
  size_t cur = 0;
  auto take = [&](size_t bytes) -> void* {
    void* p = ws + cur;
    cur += (bytes + 255) & ~(size_t)255;
    return p;
  };

  unsigned short* wqb = (unsigned short*)take((size_t)D_ * D_ * 2);
  unsigned short* wkb = (unsigned short*)take((size_t)D_ * D_ * 2);
  unsigned short* wvb = (unsigned short*)take((size_t)D_ * D_ * 2);
  unsigned short* wob = (unsigned short*)take((size_t)D_ * D_ * 2);
  unsigned short* xqb = (unsigned short*)take((size_t)BT_ * D_ * 2);
  unsigned short* xkb = (unsigned short*)take((size_t)BT_ * D_ * 2);
  unsigned short* xvb = (unsigned short*)take((size_t)BT_ * D_ * 2);
  float* qf = (float*)take((size_t)BT_ * D_ * 4);
  float* kf = (float*)take((size_t)BT_ * D_ * 4);
  float* vf = (float*)take((size_t)BT_ * D_ * 4);
  unsigned short* qn  = (unsigned short*)take((size_t)BH_ * T_ * DK_ * 2);
  unsigned short* kn  = (unsigned short*)take((size_t)BH_ * T_ * DK_ * 2);
  unsigned short* vt  = (unsigned short*)take((size_t)BH_ * DK_ * T_ * 2);
  unsigned short* ctx = (unsigned short*)take((size_t)BT_ * D_ * 2);

  const int nX = BT_ * D_;   // 3,145,728
  const int nW = D_ * D_;    // 147,456

  cvt_bf16<<<(nX + 255) / 256, 256, 0, stream>>>(query, xqb, nX);
  cvt_bf16<<<(nX + 255) / 256, 256, 0, stream>>>(key,   xkb, nX);
  cvt_bf16<<<(nX + 255) / 256, 256, 0, stream>>>(value, xvb, nX);
  cvt_bf16<<<(nW + 255) / 256, 256, 0, stream>>>(Wq, wqb, nW);
  cvt_bf16<<<(nW + 255) / 256, 256, 0, stream>>>(Wk, wkb, nW);
  cvt_bf16<<<(nW + 255) / 256, 256, 0, stream>>>(Wv, wvb, nW);
  cvt_bf16<<<(nW + 255) / 256, 256, 0, stream>>>(Wo, wob, nW);

  const int gemmBlocks = (BT_ / 16) * (D_ / 64) / 8;   // 384 blocks of 8 waves
  gemm_bf16nt<<<gemmBlocks, 256, 0, stream>>>(xqb, wqb, bq, qf, BT_);
  gemm_bf16nt<<<gemmBlocks, 256, 0, stream>>>(xkb, wkb, bk, kf, BT_);
  gemm_bf16nt<<<gemmBlocks, 256, 0, stream>>>(xvb, wvb, bv, vf, BT_);

  ln_head<<<(B_ * H_ * T_) / 8, 256, 0, stream>>>(qf, qn_w, qn_b, qn);
  ln_head<<<(B_ * H_ * T_) / 8, 256, 0, stream>>>(kf, kn_w, kn_b, kn);
  v_split<<<(nX + 255) / 256, 256, 0, stream>>>(vf, vt);

  dim3 agrid(T_ / 64, BH_);
  attn_kernel<<<agrid, 128, 0, stream>>>(qn, kn, vt, mask, ctx);

  gemm_bf16nt<<<gemmBlocks, 256, 0, stream>>>(ctx, wob, bo, out, BT_);

  (void)in_sizes; (void)n_in; (void)out_size; (void)ws_size;
}
